// BitNetBlock_63204738728234
// MI455X (gfx1250) — 16-bit and 8-bit matrix operand order repaired and checked by reading, compiled, not run on hardware
//
#include <hip/hip_runtime.h>
#include <cstdint>
#include <cstddef>

#define CB   2
#define CS   1024
#define CD   2048
#define CH   16
#define CHD  128
#define CMLP 8192
#define CM   (CB * CS)

typedef __attribute__((ext_vector_type(8)))  int      v8i;
typedef __attribute__((ext_vector_type(8)))  float    v8f;
typedef __attribute__((ext_vector_type(16))) _Float16 v16h;

union V16U { v16h h; uint32_t u[8]; };

typedef __attribute__((address_space(3))) int8_t lds_i8_t;

__device__ __forceinline__ void async_copy_b128_to_lds(void* ldsDst, const void* gSrc) {
    unsigned ldsOff = (unsigned)(uintptr_t)(lds_i8_t*)ldsDst;
    asm volatile("global_load_async_to_lds_b128 %0, %1, off"
                 :: "v"(ldsOff), "v"(gSrc)
                 : "memory");
}
__device__ __forceinline__ void wait_asynccnt0() {
    asm volatile("s_wait_asynccnt 0x0" ::: "memory");
}

__global__ void k_zero8(float* p) {
    if (threadIdx.x < 8) p[threadIdx.x] = 0.0f;
}

__global__ __launch_bounds__(256) void k_abssum(const float* __restrict__ w,
                                                size_t n, float* __restrict__ out) {
    __shared__ float red[256];
    size_t stride = (size_t)gridDim.x * blockDim.x;
    float s = 0.0f;
    for (size_t i = (size_t)blockIdx.x * blockDim.x + threadIdx.x; i < n; i += stride)
        s += fabsf(w[i]);
    red[threadIdx.x] = s;
    __syncthreads();
    for (int o = 128; o > 0; o >>= 1) {
        if ((int)threadIdx.x < o) red[threadIdx.x] += red[threadIdx.x + o];
        __syncthreads();
    }
    if (threadIdx.x == 0) atomicAdd(out, red[0]);
}

__global__ __launch_bounds__(256) void k_wquant(const float* __restrict__ w, size_t n,
                                                const float* __restrict__ wsum, float cnt,
                                                int8_t* __restrict__ out) {
    float ws = wsum[0] / cnt + 1e-8f;
    float inv = 1.0f / ws;
    size_t stride = (size_t)gridDim.x * blockDim.x;
    for (size_t i = (size_t)blockIdx.x * blockDim.x + threadIdx.x; i < n; i += stride) {
        float q = rintf(w[i] * inv);
        q = fminf(fmaxf(q, -1.0f), 1.0f);
        out[i] = (int8_t)q;
    }
}

__global__ __launch_bounds__(256) void k_rmsnorm_quant(const float* __restrict__ X,
                                                       const float* __restrict__ nw,
                                                       int8_t* __restrict__ Aq,
                                                       float* __restrict__ ascale) {
    __shared__ float red[256];
    int row = blockIdx.x;
    const float* xr = X + (size_t)row * CD;
    float loc[CD / 256];
    float ss = 0.0f;
    #pragma unroll
    for (int e = 0; e < CD / 256; ++e) {
        loc[e] = xr[threadIdx.x + e * 256];
        ss += loc[e] * loc[e];
    }
    red[threadIdx.x] = ss;
    __syncthreads();
    for (int o = 128; o > 0; o >>= 1) {
        if ((int)threadIdx.x < o) red[threadIdx.x] += red[threadIdx.x + o];
        __syncthreads();
    }
    float rinv = rsqrtf(red[0] / (float)CD + 1e-6f);
    __syncthreads();
    float amax = 0.0f;
    #pragma unroll
    for (int e = 0; e < CD / 256; ++e) {
        loc[e] = loc[e] * rinv * nw[threadIdx.x + e * 256];
        amax = fmaxf(amax, fabsf(loc[e]));
    }
    red[threadIdx.x] = amax;
    __syncthreads();
    for (int o = 128; o > 0; o >>= 1) {
        if ((int)threadIdx.x < o) red[threadIdx.x] = fmaxf(red[threadIdx.x], red[threadIdx.x + o]);
        __syncthreads();
    }
    float as = red[0] + 1e-8f;
    if (threadIdx.x == 0) ascale[row] = as;
    float sc = 127.0f / as;
    #pragma unroll
    for (int e = 0; e < CD / 256; ++e) {
        float q = rintf(fminf(fmaxf(loc[e] * sc, -128.0f), 127.0f));
        Aq[(size_t)row * CD + threadIdx.x + e * 256] = (int8_t)q;
    }
}

__global__ __launch_bounds__(256) void k_rowquant(const float* __restrict__ X, int W,
                                                  int8_t* __restrict__ Aq,
                                                  float* __restrict__ ascale) {
    __shared__ float red[256];
    int row = blockIdx.x;
    const float* xr = X + (size_t)row * W;
    float loc[32];
    int n = W / 256;
    float amax = 0.0f;
    for (int e = 0; e < n; ++e) {
        loc[e] = xr[threadIdx.x + e * 256];
        amax = fmaxf(amax, fabsf(loc[e]));
    }
    red[threadIdx.x] = amax;
    __syncthreads();
    for (int o = 128; o > 0; o >>= 1) {
        if ((int)threadIdx.x < o) red[threadIdx.x] = fmaxf(red[threadIdx.x], red[threadIdx.x + o]);
        __syncthreads();
    }
    float as = red[0] + 1e-8f;
    if (threadIdx.x == 0) ascale[row] = as;
    float sc = 127.0f / as;
    for (int e = 0; e < n; ++e) {
        float q = rintf(fminf(fmaxf(loc[e] * sc, -128.0f), 127.0f));
        Aq[(size_t)row * W + threadIdx.x + e * 256] = (int8_t)q;
    }
}

__device__ __forceinline__ v8i load_a_frag_i8(const int8_t* __restrict__ rowPtr, int hi) {
    v8i a;
    #pragma unroll
    for (int v = 0; v < 8; ++v) {
        int kb = (hi ? 8 : 0) + (v & 1) * 4 + (v >> 1) * 16;
        a[v] = *(const int*)(rowPtr + kb);
    }
    return a;
}

__device__ __forceinline__ v8i load_b_frag_i8(const int8_t* __restrict__ nRowPtr, int hi) {
    v8i b;
    #pragma unroll
    for (int v = 0; v < 8; ++v) {
        int kb = (hi ? 8 : 0) + (v & 1) * 4 + (v >> 1) * 16;
        b[v] = *(const int*)(nRowPtr + kb);
    }
    return b;
}

__global__ __launch_bounds__(256)
void k_gemm_iu8(const int8_t* __restrict__ Aq, const float* __restrict__ ascale,
                const int8_t* __restrict__ Wq, const float* __restrict__ wsum,
                float wcount, int N, int K,
                const float* __restrict__ resid, float* __restrict__ Y) {
    __shared__ __align__(16) int8_t Bt[64][64];
    int tid = threadIdx.x;
    int lane = tid & 31, lane16 = lane & 15, hi = lane >> 4;
    int wave = tid >> 5;
    int rowBase = blockIdx.x * 128 + wave * 16;
    int colBase = blockIdx.y * 64;

    v8i acc[4] = {};
    int ldr = tid >> 2;
    int ldc = (tid & 3) * 16;

    for (int k0 = 0; k0 < K; k0 += 64) {
        async_copy_b128_to_lds(&Bt[ldr][ldc],
                               Wq + (size_t)(colBase + ldr) * K + k0 + ldc);
        if (k0 + 64 < K)
            __builtin_prefetch(Wq + (size_t)(colBase + ldr) * K + k0 + 64 + ldc, 0, 1);
        wait_asynccnt0();
        __syncthreads();

        v8i a = load_a_frag_i8(Aq + (size_t)(rowBase + lane16) * K + k0, hi);
        v8i b0 = load_b_frag_i8(&Bt[0 * 16 + lane16][0], hi);
        v8i b1 = load_b_frag_i8(&Bt[1 * 16 + lane16][0], hi);
        v8i b2 = load_b_frag_i8(&Bt[2 * 16 + lane16][0], hi);
        v8i b3 = load_b_frag_i8(&Bt[3 * 16 + lane16][0], hi);
        acc[0] = __builtin_amdgcn_wmma_i32_16x16x64_iu8(true, a, true, b0, acc[0], false, false);
        acc[1] = __builtin_amdgcn_wmma_i32_16x16x64_iu8(true, a, true, b1, acc[1], false, false);
        acc[2] = __builtin_amdgcn_wmma_i32_16x16x64_iu8(true, a, true, b2, acc[2], false, false);
        acc[3] = __builtin_amdgcn_wmma_i32_16x16x64_iu8(true, a, true, b3, acc[3], false, false);
        __syncthreads();
    }

    float wscale = wsum[0] / wcount + 1e-8f;
    float as[8];
    #pragma unroll
    for (int v = 0; v < 8; ++v)
        as[v] = ascale[rowBase + hi * 8 + v] * wscale * (1.0f / 127.0f);
    #pragma unroll
    for (int j = 0; j < 4; ++j) {
        #pragma unroll
        for (int v = 0; v < 8; ++v) {
            size_t m = (size_t)(rowBase + hi * 8 + v);
            int    n = colBase + j * 16 + lane16;
            float  y = (float)acc[j][v] * as[v];
            if (resid) y += resid[m * N + n];
            Y[m * N + n] = y;
        }
    }
}

__global__ __launch_bounds__(256)
void k_gemm_iu8_gated(const int8_t* __restrict__ Aq, const float* __restrict__ ascale,
                      const int8_t* __restrict__ Wg, const int8_t* __restrict__ Wu,
                      const float* __restrict__ wsumg, const float* __restrict__ wsumu,
                      float wcount, int N, int K, float* __restrict__ Y) {
    __shared__ __align__(16) int8_t Btg[64][64];
    __shared__ __align__(16) int8_t Btu[64][64];
    int tid = threadIdx.x;
    int lane = tid & 31, lane16 = lane & 15, hi = lane >> 4;
    int wave = tid >> 5;
    int rowBase = blockIdx.x * 128 + wave * 16;
    int colBase = blockIdx.y * 64;

    v8i accg[4] = {};
    v8i accu[4] = {};
    int ldr = tid >> 2;
    int ldc = (tid & 3) * 16;

    for (int k0 = 0; k0 < K; k0 += 64) {
        async_copy_b128_to_lds(&Btg[ldr][ldc],
                               Wg + (size_t)(colBase + ldr) * K + k0 + ldc);
        async_copy_b128_to_lds(&Btu[ldr][ldc],
                               Wu + (size_t)(colBase + ldr) * K + k0 + ldc);
        wait_asynccnt0();
        __syncthreads();

        v8i a = load_a_frag_i8(Aq + (size_t)(rowBase + lane16) * K + k0, hi);
        #pragma unroll
        for (int j = 0; j < 4; ++j) {
            v8i bg = load_b_frag_i8(&Btg[j * 16 + lane16][0], hi);
            v8i bu = load_b_frag_i8(&Btu[j * 16 + lane16][0], hi);
            accg[j] = __builtin_amdgcn_wmma_i32_16x16x64_iu8(true, a, true, bg, accg[j],
                                                             false, false);
            accu[j] = __builtin_amdgcn_wmma_i32_16x16x64_iu8(true, a, true, bu, accu[j],
                                                             false, false);
        }
        __syncthreads();
    }

    float wsg = wsumg[0] / wcount + 1e-8f;
    float wsu = wsumu[0] / wcount + 1e-8f;
    #pragma unroll
    for (int j = 0; j < 4; ++j) {
        #pragma unroll
        for (int v = 0; v < 8; ++v) {
            size_t m = (size_t)(rowBase + hi * 8 + v);
            int    n = colBase + j * 16 + lane16;
            float  asm_ = ascale[m] * (1.0f / 127.0f);
            float  g = (float)accg[j][v] * wsg * asm_;
            float  u = (float)accu[j][v] * wsu * asm_;
            float  sg = g / (1.0f + __expf(-g));
            Y[m * N + n] = sg * u;
        }
    }
}

__global__ __launch_bounds__(256) void k_pack_qk(const float* __restrict__ Y,
                                                 _Float16* __restrict__ dst, float mul) {
    size_t i = (size_t)blockIdx.x * blockDim.x + threadIdx.x;
    int m = (int)(i / CD), d = (int)(i % CD);
    int b = m / CS, s = m % CS, h = d / CHD, hd = d % CHD;
    dst[(((size_t)(b * CH + h) * CS + s) * CHD) + hd] = (_Float16)(Y[i] * mul);
}

__global__ __launch_bounds__(256) void k_pack_vT(const float* __restrict__ Y,
                                                 _Float16* __restrict__ dst) {
    size_t i = (size_t)blockIdx.x * blockDim.x + threadIdx.x;
    int m = (int)(i / CD), d = (int)(i % CD);
    int b = m / CS, s = m % CS, h = d / CHD, hd = d % CHD;
    dst[(((size_t)(b * CH + h) * CHD + hd) * CS) + s] = (_Float16)Y[i];
}

__global__ __launch_bounds__(32)
void k_flash_attn(const _Float16* __restrict__ Qh, const _Float16* __restrict__ Kh,
                  const _Float16* __restrict__ Vt, float* __restrict__ Ctx) {
    __shared__ __align__(16) _Float16 Pl[16][32];
    int lane = threadIdx.x;
    int lane16 = lane & 15, hi = lane >> 4;
    int q0 = blockIdx.x * 16;
    int bh = blockIdx.y;
    int b = bh / CH, h = bh % CH;

    const _Float16* Qb = Qh + (size_t)bh * CS * CHD;
    const _Float16* Kb = Kh + (size_t)bh * CS * CHD;
    const _Float16* Vb = Vt + (size_t)bh * CHD * CS;

    V16U qf[4];
    #pragma unroll
    for (int c = 0; c < 4; ++c)
        #pragma unroll
        for (int v = 0; v < 8; ++v) {
            int kh = (hi ? 16 : 0) + v * 2;
            qf[c].u[v] = *(const uint32_t*)(Qb + (size_t)(q0 + lane16) * CHD + c * 32 + kh);
        }

    v8f out[8] = {};
    float mrun[8], lrun[8];
    #pragma unroll
    for (int v = 0; v < 8; ++v) { mrun[v] = -1e30f; lrun[v] = 0.0f; }

    int nsteps = q0 / 32 + 1;
    for (int st = 0; st < nsteps; ++st) {
        int kb = st * 32;
        float p[2][8], mnew[8];
        #pragma unroll
        for (int v = 0; v < 8; ++v) mnew[v] = mrun[v];

        #pragma unroll
        for (int t = 0; t < 2; ++t) {
            v8f sacc = {};
            #pragma unroll
            for (int c = 0; c < 4; ++c) {
                V16U kf;
                #pragma unroll
                for (int v = 0; v < 8; ++v) {
                    int kh = (hi ? 16 : 0) + v * 2;
                    kf.u[v] = *(const uint32_t*)(Kb + (size_t)(kb + t * 16 + lane16) * CHD
                                                 + c * 32 + kh);
                }
                sacc = __builtin_amdgcn_wmma_f32_16x16x32_f16(false, qf[c].h, false, kf.h,
                                                              (short)0, sacc, false, false);
            }
            int key = kb + t * 16 + lane16;
            #pragma unroll
            for (int v = 0; v < 8; ++v) {
                int row = q0 + hi * 8 + v;
                p[t][v] = (key > row) ? -1e30f : sacc[v];
            }
        }

        #pragma unroll
        for (int v = 0; v < 8; ++v) {
            float mx = fmaxf(p[0][v], p[1][v]);
            #pragma unroll
            for (int off = 1; off < 16; off <<= 1)
                mx = fmaxf(mx, __shfl_xor(mx, off, 32));
            mnew[v] = fmaxf(mnew[v], mx);
        }
        float rsum[8];
        #pragma unroll
        for (int v = 0; v < 8; ++v) {
            float e0 = __expf(p[0][v] - mnew[v]);
            float e1 = __expf(p[1][v] - mnew[v]);
            p[0][v] = e0; p[1][v] = e1;
            float sm = e0 + e1;
            #pragma unroll
            for (int off = 1; off < 16; off <<= 1)
                sm += __shfl_xor(sm, off, 32);
            rsum[v] = sm;
        }
        #pragma unroll
        for (int v = 0; v < 8; ++v) {
            float r = __expf(mrun[v] - mnew[v]);
            lrun[v] = lrun[v] * r + rsum[v];
            mrun[v] = mnew[v];
            #pragma unroll
            for (int dd = 0; dd < 8; ++dd) out[dd][v] *= r;
        }

        #pragma unroll
        for (int v = 0; v < 8; ++v) {
            Pl[hi * 8 + v][lane16]      = (_Float16)p[0][v];
            Pl[hi * 8 + v][16 + lane16] = (_Float16)p[1][v];
        }
        __syncthreads();
        V16U pa;
        #pragma unroll
        for (int v = 0; v < 8; ++v) {
            int kh = (hi ? 16 : 0) + v * 2;
            pa.u[v] = *(const uint32_t*)(&Pl[lane16][kh]);
        }
        #pragma unroll
        for (int dd = 0; dd < 8; ++dd) {
            V16U vf;
            #pragma unroll
            for (int v = 0; v < 8; ++v) {
                int nn = (hi ? 16 : 0) + v * 2;
                vf.u[v] = *(const uint32_t*)(Vb + (size_t)(dd * 16 + lane16) * CS + kb + nn);
            }
            out[dd] = __builtin_amdgcn_wmma_f32_16x16x32_f16(false, pa.h, false, vf.h,
                                                             (short)0, out[dd], false, false);
        }
        __syncthreads();
    }

    #pragma unroll
    for (int dd = 0; dd < 8; ++dd)
        #pragma unroll
        for (int v = 0; v < 8; ++v) {
            int row = q0 + hi * 8 + v;
            int col = h * CHD + dd * 16 + lane16;
            Ctx[((size_t)(b * CS + row)) * CD + col] = out[dd][v] / lrun[v];
        }
}

extern "C" void kernel_launch(void* const* d_in, const int* in_sizes, int n_in,
                              void* d_out, int out_size, void* d_ws, size_t ws_size,
                              hipStream_t stream) {
    const float* x     = (const float*)d_in[0];
    const float* wq    = (const float*)d_in[1];
    const float* wk    = (const float*)d_in[2];
    const float* wv    = (const float*)d_in[3];
    const float* wo    = (const float*)d_in[4];
    const float* wg    = (const float*)d_in[5];
    const float* wu    = (const float*)d_in[6];
    const float* wd    = (const float*)d_in[7];
    const float* norm1 = (const float*)d_in[8];
    const float* norm2 = (const float*)d_in[9];

    const size_t DDn = (size_t)CD * CD;
    const size_t DMn = (size_t)CD * CMLP;

    uint8_t* ws = (uint8_t*)d_ws;
    size_t o = 0;
    float*    wsum   = (float*)(ws + o);  o += 256;
    float*    ascale = (float*)(ws + o);  o += (size_t)CM * 4;
    o = (o + 255) & ~(size_t)255;
    int8_t*   wqq = (int8_t*)(ws + o);    o += DDn;
    int8_t*   wkq = (int8_t*)(ws + o);    o += DDn;
    int8_t*   wvq = (int8_t*)(ws + o);    o += DDn;
    int8_t*   woq = (int8_t*)(ws + o);    o += DDn;
    int8_t*   wgq = (int8_t*)(ws + o);    o += DMn;
    int8_t*   wuq = (int8_t*)(ws + o);    o += DMn;
    int8_t*   wdq = (int8_t*)(ws + o);    o += DMn;
    int8_t*   aq  = (int8_t*)(ws + o);    o += (size_t)CM * CMLP;
    _Float16* qh  = (_Float16*)(ws + o);  o += (size_t)CM * CD * 2;
    _Float16* kh  = (_Float16*)(ws + o);  o += (size_t)CM * CD * 2;
    _Float16* vt  = (_Float16*)(ws + o);  o += (size_t)CM * CD * 2;
    float*    Ybuf = (float*)(ws + o);    o += (size_t)CM * CMLP * 4;
    float*    x1   = (float*)(ws + o);    o += (size_t)CM * CD * 4;
    (void)ws_size; (void)in_sizes; (void)n_in; (void)out_size;

    k_zero8<<<1, 32, 0, stream>>>(wsum);
    const float* Wsrc[7] = {wq, wk, wv, wo, wg, wu, wd};
    size_t       Wn[7]   = {DDn, DDn, DDn, DDn, DMn, DMn, DMn};
    int8_t*      Wq8[7]  = {wqq, wkq, wvq, woq, wgq, wuq, wdq};
    for (int i = 0; i < 7; ++i)
        k_abssum<<<512, 256, 0, stream>>>(Wsrc[i], Wn[i], wsum + i);
    for (int i = 0; i < 7; ++i)
        k_wquant<<<2048, 256, 0, stream>>>(Wsrc[i], Wn[i], wsum + i, (float)Wn[i], Wq8[i]);

    const dim3 gemmD(CM / 128, CD / 64);
    const dim3 gemmM(CM / 128, CMLP / 64);
    const int  packBlocks = (int)(((size_t)CM * CD) / 256);
    const float invSqrtHD = 0.088388347648318447f;

    k_rmsnorm_quant<<<CM, 256, 0, stream>>>(x, norm1, aq, ascale);
    k_gemm_iu8<<<gemmD, 256, 0, stream>>>(aq, ascale, wqq, wsum + 0, (float)DDn,
                                          CD, CD, nullptr, Ybuf);
    k_pack_qk<<<packBlocks, 256, 0, stream>>>(Ybuf, qh, invSqrtHD);
    k_gemm_iu8<<<gemmD, 256, 0, stream>>>(aq, ascale, wkq, wsum + 1, (float)DDn,
                                          CD, CD, nullptr, Ybuf);
    k_pack_qk<<<packBlocks, 256, 0, stream>>>(Ybuf, kh, 1.0f);
    k_gemm_iu8<<<gemmD, 256, 0, stream>>>(aq, ascale, wvq, wsum + 2, (float)DDn,
                                          CD, CD, nullptr, Ybuf);
    k_pack_vT<<<packBlocks, 256, 0, stream>>>(Ybuf, vt);

    k_flash_attn<<<dim3(CS / 16, CB * CH), 32, 0, stream>>>(qh, kh, vt, Ybuf);

    k_rowquant<<<CM, 256, 0, stream>>>(Ybuf, CD, aq, ascale);
    k_gemm_iu8<<<gemmD, 256, 0, stream>>>(aq, ascale, woq, wsum + 3, (float)DDn,
                                          CD, CD, x, x1);

    k_rmsnorm_quant<<<CM, 256, 0, stream>>>(x1, norm2, aq, ascale);
    k_gemm_iu8_gated<<<gemmM, 256, 0, stream>>>(aq, ascale, wgq, wuq, wsum + 4, wsum + 5,
                                                (float)DMn, CMLP, CD, Ybuf);
    k_rowquant<<<CM, 256, 0, stream>>>(Ybuf, CMLP, aq, ascale);
    k_gemm_iu8<<<gemmD, 256, 0, stream>>>(aq, ascale, wdq, wsum + 6, (float)DMn,
                                          CD, CMLP, x1, (float*)d_out);
}
